// EventGroupSpecificMMoE_85907935855400
// MI455X (gfx1250) — matrix operand order repaired and checked by reading, not compiled, not run on hardware
//
#include <hip/hip_runtime.h>

// ---- problem dims ----
#define N_TOK 4096
#define D_DIM 512
#define H_DIM 1024
#define S_EXP 2
#define G_EXP 2
#define NG_   6
#define C_DIM 32
#define K_TASK 3
#define NEV_  50
#define EFF   (D_DIM + C_DIM)   // 544
#define E_SLOTS 4

typedef __attribute__((ext_vector_type(16))) __bf16 bf16x16;
typedef __attribute__((ext_vector_type(8)))  __bf16 bf16x8;
typedef __attribute__((ext_vector_type(4)))  __bf16 bf16x4;
typedef __attribute__((ext_vector_type(8)))  float  f32x8;

static __device__ __forceinline__ __bf16 f2bf(float f) {
  union { float f; unsigned u; } a; a.f = f;
  unsigned r = a.u + 0x7FFFu + ((a.u >> 16) & 1u);
  unsigned short s = (unsigned short)(r >> 16);
  __bf16 o; __builtin_memcpy(&o, &s, 2); return o;
}

// ---------- prep: group ids + per-block histograms ----------
__global__ void k_gids(const int* __restrict__ ntid, const int* __restrict__ e2g,
                       int* __restrict__ gids, int* __restrict__ blockHist) {
  __shared__ int h[NG_];
  if (threadIdx.x < NG_) h[threadIdx.x] = 0;
  __syncthreads();
  int t = blockIdx.x * 256 + threadIdx.x;
  int v = ntid[t];
  v = v < 0 ? 0 : (v > NEV_ - 1 ? NEV_ - 1 : v);
  int g = e2g[v];
  gids[t] = g;
  atomicAdd(&h[g], 1);
  __syncthreads();
  if (threadIdx.x < NG_) blockHist[blockIdx.x * NG_ + threadIdx.x] = h[threadIdx.x];
}

// ---------- prep: scan histograms (single tiny block) ----------
__global__ void k_scan(const int* __restrict__ blockHist, int* __restrict__ counts,
                       int* __restrict__ groupBase, int* __restrict__ blockBase) {
  __shared__ int cnt[NG_], gb[NG_];
  int g = threadIdx.x;
  if (g < NG_) {
    int s = 0;
    for (int b = 0; b < 16; ++b) s += blockHist[b * NG_ + g];
    cnt[g] = s; counts[g] = s;
  }
  __syncthreads();
  if (threadIdx.x == 0) {
    int a = 0;
    for (int g2 = 0; g2 < NG_; ++g2) { gb[g2] = a; a += cnt[g2]; }
  }
  __syncthreads();
  if (g < NG_) {
    groupBase[g] = gb[g];
    int run = gb[g];
    for (int b = 0; b < 16; ++b) { blockBase[b * NG_ + g] = run; run += blockHist[b * NG_ + g]; }
  }
}

// ---------- prep: stable rank scatter ----------
__global__ void k_perm(const int* __restrict__ gids, const int* __restrict__ blockBase,
                       int* __restrict__ perm) {
  __shared__ int lg[256];
  int t = blockIdx.x * 256 + threadIdx.x;
  int g = gids[t];
  lg[threadIdx.x] = g;
  __syncthreads();
  int r = blockBase[blockIdx.x * NG_ + g];
  for (int j = 0; j < (int)threadIdx.x; ++j) r += (lg[j] == g);
  perm[r] = t;
}

// ---------- prep: x -> bf16 ----------
__global__ void k_xb(const float* __restrict__ x, __bf16* __restrict__ xb) {
  int id = blockIdx.x * 256 + threadIdx.x;          // 2048 blocks * 256 * 4 = 2M elems
  float4 v = ((const float4*)x)[id];
  bf16x4 o = { f2bf(v.x), f2bf(v.y), f2bf(v.z), f2bf(v.w) };
  ((bf16x4*)xb)[id] = o;
}

// ---------- prep: fp32 [E][R][C] -> bf16 transposed [E][C][R] ----------
__global__ void k_tr(const float* __restrict__ in, __bf16* __restrict__ out, int R, int Cc) {
  __shared__ float tile[32][33];
  int e = blockIdx.z;
  const float* ip = in + (size_t)e * R * Cc;
  __bf16* op = out + (size_t)e * R * Cc;
  int c0 = blockIdx.x * 32, r0 = blockIdx.y * 32;
  for (int i = 0; i < 32; i += 8)
    tile[threadIdx.y + i][threadIdx.x] = ip[(size_t)(r0 + threadIdx.y + i) * Cc + c0 + threadIdx.x];
  __syncthreads();
  for (int i = 0; i < 32; i += 8)
    op[(size_t)(c0 + threadIdx.y + i) * R + r0 + threadIdx.x] = f2bf(tile[threadIdx.x][threadIdx.y + i]);
}

// ---------- gate logits + softmax (fp32 exact) ----------
__global__ void k_gate(const float* __restrict__ x, const int* __restrict__ gids,
                       const float* __restrict__ cond_emb, const float* __restrict__ gate_W,
                       const float* __restrict__ gate_b, float* __restrict__ gatew) {
  int id = blockIdx.x * 256 + threadIdx.x;          // 48*256 == N_TOK*K_TASK exactly
  int t = id / K_TASK, k = id - t * K_TASK;
  int g = gids[t];
  float a0 = 0.f, a1 = 0.f, a2 = 0.f, a3 = 0.f;
  const float* xr = x + (size_t)t * D_DIM;
  const float* wk = gate_W + (size_t)k * EFF * E_SLOTS;
  for (int f = 0; f < EFF; ++f) {
    float xv = (f < D_DIM) ? xr[f] : cond_emb[g * C_DIM + (f - D_DIM)];
    float4 w = *(const float4*)(wk + f * 4);
    a0 += xv * w.x; a1 += xv * w.y; a2 += xv * w.z; a3 += xv * w.w;
  }
  a0 += gate_b[k * 4 + 0]; a1 += gate_b[k * 4 + 1];
  a2 += gate_b[k * 4 + 2]; a3 += gate_b[k * 4 + 3];
  float m = fmaxf(fmaxf(a0, a1), fmaxf(a2, a3));
  float e0 = expf(a0 - m), e1 = expf(a1 - m), e2 = expf(a2 - m), e3 = expf(a3 - m);
  float inv = 1.f / (e0 + e1 + e2 + e3);
  float* o = gatew + (size_t)t * (K_TASK * E_SLOTS) + k * E_SLOTS;
  o[0] = e0 * inv; o[1] = e1 * inv; o[2] = e2 * inv; o[3] = e3 * inv;
}

// ---------- fragment helpers ----------
static __device__ __forceinline__ bf16x16 frag_A(const __bf16* base, int stride, int nlo, int hi) {
  // A 16x32: lane = row; K chunks at k0+8*hi and k0+8*hi+16
  const __bf16* p = base + nlo * stride + 8 * hi;
  bf16x8 lo = *(const bf16x8*)p;
  bf16x8 h8 = *(const bf16x8*)(p + 16);
  return __builtin_shufflevector(lo, h8, 0,1,2,3,4,5,6,7,8,9,10,11,12,13,14,15);
}
static __device__ __forceinline__ bf16x16 frag_B(const __bf16* base, int stride, int nlo, int hi) {
  // B 32x16: lane = col; K chunks at k0+8*hi and k0+8*hi+16
  const __bf16* p = base + (size_t)nlo * stride + 8 * hi;
  bf16x8 lo = *(const bf16x8*)p;
  bf16x8 h8 = *(const bf16x8*)(p + 16);
  return __builtin_shufflevector(lo, h8, 0,1,2,3,4,5,6,7,8,9,10,11,12,13,14,15);
}

#define SA 520   // LDS row stride (elems) for X tile: 1040B -> 16B rotation mod 256B banks
#define SH 1032  // LDS row stride (elems) for H tile: 2064B -> 16B rotation

// ---------- fused expert FFN: out[64,512] = relu(x[64,512]@W1+b1)@W2+b2 ----------
__global__ void __launch_bounds__(256, 1)
k_expert(const __bf16* __restrict__ xb,
         const __bf16* __restrict__ Ws1t, const float* __restrict__ bs1,
         const __bf16* __restrict__ Ws2t, const float* __restrict__ bs2,
         const __bf16* __restrict__ Wg1t, const float* __restrict__ bg1,
         const __bf16* __restrict__ Wg2t, const float* __restrict__ bg2,
         const int* __restrict__ counts, const int* __restrict__ groupBase,
         const int* __restrict__ perm, float* __restrict__ eo) {
  extern __shared__ char smem[];
  __bf16* XA = (__bf16*)smem;              // [64][SA]
  __bf16* HL = XA + 64 * SA;               // [64][SH]
  int* tokrow = (int*)(HL + 64 * SH);      // [64]

  const int y = blockIdx.y;
  const __bf16 *W1, *W2;
  const float *b1, *b2;
  int eslot;
  if (y < S_EXP) {
    W1 = Ws1t + (size_t)y * H_DIM * D_DIM; b1 = bs1 + y * H_DIM;
    W2 = Ws2t + (size_t)y * D_DIM * H_DIM; b2 = bs2 + y * D_DIM;
    eslot = y;
    if (threadIdx.x < 64) tokrow[threadIdx.x] = blockIdx.x * 64 + threadIdx.x;
  } else {
    int bi = y - S_EXP;
    int ng = bi >> 1, g = bi & 1;
    int cnt = counts[ng];
    int start = blockIdx.x * 64;
    if (start >= cnt) return;              // uniform exit, no barrier taken yet
    W1 = Wg1t + (size_t)bi * H_DIM * D_DIM; b1 = bg1 + bi * H_DIM;
    W2 = Wg2t + (size_t)bi * D_DIM * H_DIM; b2 = bg2 + bi * D_DIM;
    eslot = S_EXP + g;
    int base = groupBase[ng];
    if (threadIdx.x < 64) {
      int r = start + threadIdx.x;
      tokrow[threadIdx.x] = (r < cnt) ? perm[base + r] : -1;
    }
  }
  __syncthreads();

  // stage X tile (64 x 512 bf16) into LDS, 16B chunks
  for (int c = threadIdx.x; c < 64 * 64; c += 256) {
    int row = c >> 6, off = (c & 63) * 8;
    int tok = tokrow[row]; if (tok < 0) tok = 0;
    *(bf16x8*)(XA + row * SA + off) = *(const bf16x8*)(xb + (size_t)tok * D_DIM + off);
  }
  __syncthreads();

  const int lane = threadIdx.x & 31;
  const int wv   = threadIdx.x >> 5;
  const int nlo  = lane & 15;
  const int hi   = lane >> 4;
  const f32x8 zero = {0.f,0.f,0.f,0.f,0.f,0.f,0.f,0.f};

  // ---- stage 1: H = relu(X @ W1 + b1), each wave owns 128 hidden cols ----
  for (int ct = 0; ct < 8; ++ct) {
    const int n0 = wv * 128 + ct * 16;
    f32x8 acc0 = zero, acc1 = zero, acc2 = zero, acc3 = zero;
    for (int kb = 0; kb < 16; ++kb) {
      const int k0 = kb * 32;
      bf16x16 bfr = frag_B(W1 + (size_t)n0 * D_DIM + k0, D_DIM, nlo, hi);
      bf16x16 a0 = frag_A(XA + 0  * SA + k0, SA, nlo, hi);
      bf16x16 a1 = frag_A(XA + 16 * SA + k0, SA, nlo, hi);
      bf16x16 a2 = frag_A(XA + 32 * SA + k0, SA, nlo, hi);
      bf16x16 a3 = frag_A(XA + 48 * SA + k0, SA, nlo, hi);
      acc0 = __builtin_amdgcn_wmma_f32_16x16x32_bf16(false, a0, false, bfr, (short)0, acc0, false, false);
      acc1 = __builtin_amdgcn_wmma_f32_16x16x32_bf16(false, a1, false, bfr, (short)0, acc1, false, false);
      acc2 = __builtin_amdgcn_wmma_f32_16x16x32_bf16(false, a2, false, bfr, (short)0, acc2, false, false);
      acc3 = __builtin_amdgcn_wmma_f32_16x16x32_bf16(false, a3, false, bfr, (short)0, acc3, false, false);
    }
    float bv = b1[n0 + nlo];
    #pragma unroll
    for (int j = 0; j < 8; ++j) {
      int rbase = j + 8 * hi;
      HL[(0  + rbase) * SH + n0 + nlo] = f2bf(fmaxf(acc0[j] + bv, 0.f));
      HL[(16 + rbase) * SH + n0 + nlo] = f2bf(fmaxf(acc1[j] + bv, 0.f));
      HL[(32 + rbase) * SH + n0 + nlo] = f2bf(fmaxf(acc2[j] + bv, 0.f));
      HL[(48 + rbase) * SH + n0 + nlo] = f2bf(fmaxf(acc3[j] + bv, 0.f));
    }
  }
  __syncthreads();

  // ---- stage 2: OUT = H @ W2 + b2, each wave owns 64 output cols ----
  for (int ct = 0; ct < 4; ++ct) {
    const int n0 = wv * 64 + ct * 16;
    f32x8 acc0 = zero, acc1 = zero, acc2 = zero, acc3 = zero;
    for (int kb = 0; kb < 32; ++kb) {
      const int k0 = kb * 32;
      bf16x16 bfr = frag_B(W2 + (size_t)n0 * H_DIM + k0, H_DIM, nlo, hi);
      bf16x16 a0 = frag_A(HL + 0  * SH + k0, SH, nlo, hi);
      bf16x16 a1 = frag_A(HL + 16 * SH + k0, SH, nlo, hi);
      bf16x16 a2 = frag_A(HL + 32 * SH + k0, SH, nlo, hi);
      bf16x16 a3 = frag_A(HL + 48 * SH + k0, SH, nlo, hi);
      acc0 = __builtin_amdgcn_wmma_f32_16x16x32_bf16(false, a0, false, bfr, (short)0, acc0, false, false);
      acc1 = __builtin_amdgcn_wmma_f32_16x16x32_bf16(false, a1, false, bfr, (short)0, acc1, false, false);
      acc2 = __builtin_amdgcn_wmma_f32_16x16x32_bf16(false, a2, false, bfr, (short)0, acc2, false, false);
      acc3 = __builtin_amdgcn_wmma_f32_16x16x32_bf16(false, a3, false, bfr, (short)0, acc3, false, false);
    }
    float bv = b2[n0 + nlo];
    #pragma unroll
    for (int j = 0; j < 8; ++j) {
      int rbase = j + 8 * hi;
      float v0 = acc0[j] + bv, v1 = acc1[j] + bv, v2 = acc2[j] + bv, v3 = acc3[j] + bv;
      int t0 = tokrow[0 + rbase], t1 = tokrow[16 + rbase], t2 = tokrow[32 + rbase], t3 = tokrow[48 + rbase];
      if (t0 >= 0) eo[((size_t)t0 * E_SLOTS + eslot) * D_DIM + n0 + nlo] = v0;
      if (t1 >= 0) eo[((size_t)t1 * E_SLOTS + eslot) * D_DIM + n0 + nlo] = v1;
      if (t2 >= 0) eo[((size_t)t2 * E_SLOTS + eslot) * D_DIM + n0 + nlo] = v2;
      if (t3 >= 0) eo[((size_t)t3 * E_SLOTS + eslot) * D_DIM + n0 + nlo] = v3;
    }
  }
}

// ---------- final gated combine: out[k,t,d] = sum_e w[t,k,e]*eo[t,e,d] ----------
__global__ void k_combine(const float* __restrict__ eo, const float* __restrict__ gatew,
                          float* __restrict__ out) {
  int id = blockIdx.x * 256 + threadIdx.x;   // 8192 blocks -> 2M = N*D
  int t = id >> 9, d = id & 511;
  const float* er = eo + (size_t)t * E_SLOTS * D_DIM + d;
  float e0 = er[0], e1 = er[D_DIM], e2 = er[2 * D_DIM], e3 = er[3 * D_DIM];
  const float* w = gatew + (size_t)t * (K_TASK * E_SLOTS);
  #pragma unroll
  for (int k = 0; k < K_TASK; ++k) {
    float v = w[k*4+0]*e0 + w[k*4+1]*e1 + w[k*4+2]*e2 + w[k*4+3]*e3;
    out[(size_t)k * N_TOK * D_DIM + (size_t)t * D_DIM + d] = v;
  }
}

extern "C" void kernel_launch(void* const* d_in, const int* in_sizes, int n_in,
                              void* d_out, int out_size, void* d_ws, size_t ws_size,
                              hipStream_t stream) {
  const float* x    = (const float*)d_in[0];
  const int* ntid   = (const int*)d_in[1];
  const int* e2g    = (const int*)d_in[2];
  const float* Ws1  = (const float*)d_in[3];
  const float* bs1  = (const float*)d_in[4];
  const float* Ws2  = (const float*)d_in[5];
  const float* bs2  = (const float*)d_in[6];
  const float* Wg1  = (const float*)d_in[7];
  const float* bg1  = (const float*)d_in[8];
  const float* Wg2  = (const float*)d_in[9];
  const float* bg2  = (const float*)d_in[10];
  const float* cemb = (const float*)d_in[11];
  const float* gW   = (const float*)d_in[12];
  const float* gb   = (const float*)d_in[13];
  float* out = (float*)d_out;

  char* ws = (char*)d_ws;
  size_t o = 0;
  auto carve = [&](size_t bytes) { void* p = ws + o; o = (o + bytes + 255) & ~(size_t)255; return p; };
  __bf16* xb    = (__bf16*)carve((size_t)N_TOK * D_DIM * 2);
  __bf16* Ws1t  = (__bf16*)carve((size_t)S_EXP * H_DIM * D_DIM * 2);
  __bf16* Wg1t  = (__bf16*)carve((size_t)NG_ * G_EXP * H_DIM * D_DIM * 2);
  __bf16* Ws2t  = (__bf16*)carve((size_t)S_EXP * D_DIM * H_DIM * 2);
  __bf16* Wg2t  = (__bf16*)carve((size_t)NG_ * G_EXP * D_DIM * H_DIM * 2);
  float*  eo    = (float*)carve((size_t)N_TOK * E_SLOTS * D_DIM * 4);
  float*  gatew = (float*)carve((size_t)N_TOK * K_TASK * E_SLOTS * 4);
  int* gids      = (int*)carve(N_TOK * 4);
  int* perm      = (int*)carve(N_TOK * 4);
  int* blockHist = (int*)carve(16 * NG_ * 4);
  int* blockBase = (int*)carve(16 * NG_ * 4);
  int* counts    = (int*)carve(NG_ * 4);
  int* groupBase = (int*)carve(NG_ * 4);

  // token bucketing (deterministic)
  k_gids<<<16, 256, 0, stream>>>(ntid, e2g, gids, blockHist);
  k_scan<<<1, 32, 0, stream>>>(blockHist, counts, groupBase, blockBase);
  k_perm<<<16, 256, 0, stream>>>(gids, blockBase, perm);

  // precision/layout prep
  k_xb<<<2048, 256, 0, stream>>>(x, xb);
  k_tr<<<dim3(H_DIM/32, D_DIM/32, S_EXP),        dim3(32, 8), 0, stream>>>(Ws1, Ws1t, D_DIM, H_DIM);
  k_tr<<<dim3(H_DIM/32, D_DIM/32, NG_*G_EXP),    dim3(32, 8), 0, stream>>>(Wg1, Wg1t, D_DIM, H_DIM);
  k_tr<<<dim3(D_DIM/32, H_DIM/32, S_EXP),        dim3(32, 8), 0, stream>>>(Ws2, Ws2t, H_DIM, D_DIM);
  k_tr<<<dim3(D_DIM/32, H_DIM/32, NG_*G_EXP),    dim3(32, 8), 0, stream>>>(Wg2, Wg2t, H_DIM, D_DIM);

  // gates
  k_gate<<<48, 256, 0, stream>>>(x, gids, cemb, gW, gb, gatew);

  // fused expert FFNs (WMMA): y<2 shared experts over all tokens; y>=2 bucketed group banks
  size_t smem = (size_t)(64 * SA + 64 * SH) * 2 + 64 * 4;
  k_expert<<<dim3(64, S_EXP + NG_ * G_EXP), 256, smem, stream>>>(
      xb, Ws1t, bs1, Ws2t, bs2, Wg1t, bg1, Wg2t, bg2, counts, groupBase, perm, eo);

  // gated combine
  k_combine<<<8192, 256, 0, stream>>>(eo, gatew, out);
}
